// ViGBlock_22814866276970
// MI455X (gfx1250) — matrix operand order repaired and checked by reading, not compiled, not run on hardware
//
#include <hip/hip_runtime.h>
#include <cstdint>
#include <cstddef>

// ---------------------------------------------------------------------------
// Types for CDNA5 WMMA (wave32)
// ---------------------------------------------------------------------------
typedef __bf16 bf16;
typedef __attribute__((ext_vector_type(16))) __bf16 v16bf;
typedef __attribute__((ext_vector_type(8)))  __bf16 v8bf;
typedef __attribute__((ext_vector_type(8)))  float  v8f;

#define BN_EPS 1e-5f

// ---------------------------------------------------------------------------
// Fragment loaders (16-bit A 16x32 / B 32x16, contiguous k order used for both A and B)
// A, lane L: row M = L%16, K range (L/16)*16..+15 (same order as B).
//   -> two contiguous 16B chunks at p and p+8 elements.
// B, lane L: column N = L%16, K range (L/16)*16..+15 -> 16 contiguous bf16.
// ---------------------------------------------------------------------------
__device__ __forceinline__ v16bf ld_frag_a(const bf16* __restrict__ p) {
    v8bf lo = *(const v8bf*)(p);
    v8bf hi = *(const v8bf*)(p + 8);
    return __builtin_shufflevector(lo, hi, 0,1,2,3,4,5,6,7,8,9,10,11,12,13,14,15);
}
__device__ __forceinline__ v16bf ld_frag_b(const bf16* __restrict__ p) {
    v8bf lo = *(const v8bf*)(p);
    v8bf hi = *(const v8bf*)(p + 8);
    return __builtin_shufflevector(lo, hi, 0,1,2,3,4,5,6,7,8,9,10,11,12,13,14,15);
}
__device__ __forceinline__ v8f wmma_bf16(v16bf a, v16bf b, v8f c) {
    return __builtin_amdgcn_wmma_f32_16x16x32_bf16(
        /*neg_a=*/false, a, /*neg_b=*/false, b,
        /*c_mod=*/(short)0, c, /*reuse_a=*/false, /*reuse_b=*/false);
}
__device__ __forceinline__ float gelu_exact(float x) {
    return 0.5f * x * (1.0f + erff(x * 0.70710678118654752440f));
}

// ---------------------------------------------------------------------------
// Utility kernels
// ---------------------------------------------------------------------------
__global__ void k_zero_f32(float* __restrict__ p, size_t n) {
    size_t i = (size_t)blockIdx.x * blockDim.x + threadIdx.x;
    size_t stride = (size_t)gridDim.x * blockDim.x;
    for (; i < n; i += stride) p[i] = 0.0f;
}

__global__ void k_f32_to_bf16(const float* __restrict__ src, bf16* __restrict__ dst, size_t n) {
    size_t i = (size_t)blockIdx.x * blockDim.x + threadIdx.x;
    if (i < n) dst[i] = (bf16)src[i];
}

// W[K,N] f32 (row-major) -> Wt[N,K] bf16 (row-major); weights are tiny.
__global__ void k_transpose_w(const float* __restrict__ W, bf16* __restrict__ Wt, int K, int N) {
    int i = blockIdx.x * blockDim.x + threadIdx.x;
    if (i >= K * N) return;
    int k = i / N, n = i - k * N;
    Wt[(size_t)n * K + k] = (bf16)W[i];
}

// ---------------------------------------------------------------------------
// WMMA GEMM:  Y[M,Nc] (f32) = A[M,K] (bf16) @ Wt[Nc,K]^T (bf16) + bias
// One wave -> 16x64 output tile (4 accumulators, A frag reused 4x).
// Requires: M%16==0, Nc%64==0, K%32==0 (true here: M=100000, Nc in {192,256,512},
// K in {192,256,512}).
// ---------------------------------------------------------------------------
__global__ __launch_bounds__(256) void k_gemm_bias_f32(
    const bf16* __restrict__ A, const bf16* __restrict__ Wt,
    const float* __restrict__ bias, float* __restrict__ Y,
    int M, int Nc, int K)
{
    const int lane = threadIdx.x & 31;
    const int wv   = blockIdx.x * (blockDim.x >> 5) + (threadIdx.x >> 5);
    const int nG = Nc >> 6, mT = M >> 4;
    if (wv >= mT * nG) return;
    const int m0 = (wv / nG) << 4;
    const int n0 = (wv % nG) << 6;
    const int colL = lane & 15;
    const int row  = m0 + colL;
    const int kA = (lane >> 4) << 4;   // 0 or 16
    const int kB = (lane >> 4) << 4;   // 0 or 16

    const bf16* ap  = A  + (size_t)row * K + kA;
    const bf16* bp  = Wt + (size_t)(n0 + colL) * K + kB;

    v8f acc0 = {}, acc1 = {}, acc2 = {}, acc3 = {};
    for (int kk = 0; kk < K; kk += 32) {
        v16bf a = ld_frag_a(ap + kk);
        acc0 = wmma_bf16(a, ld_frag_b(bp + (size_t)(0 * 16) * K + kk), acc0);
        acc1 = wmma_bf16(a, ld_frag_b(bp + (size_t)(1 * 16) * K + kk), acc1);
        acc2 = wmma_bf16(a, ld_frag_b(bp + (size_t)(2 * 16) * K + kk), acc2);
        acc3 = wmma_bf16(a, ld_frag_b(bp + (size_t)(3 * 16) * K + kk), acc3);
    }
    // C/D layout: VGPR v, lanes 0-15 -> M=v, lanes 16-31 -> M=v+8; N=lane%16.
    const int rbase = m0 + ((lane >> 4) << 3);
#pragma unroll
    for (int v = 0; v < 8; ++v) {
        const size_t r = (size_t)(rbase + v) * Nc;
        Y[r + n0 +  0 + colL] = acc0[v] + bias[n0 +  0 + colL];
        Y[r + n0 + 16 + colL] = acc1[v] + bias[n0 + 16 + colL];
        Y[r + n0 + 32 + colL] = acc2[v] + bias[n0 + 32 + colL];
        Y[r + n0 + 48 + colL] = acc3[v] + bias[n0 + 48 + colL];
    }
}

// Dual GEMM with fused GELU epilogue, bf16 output:
//   G = gelu(A1@Wt1^T + A2@Wt2^T + bias)   (GraphConv: agg@Wrel + h@Wroot + brel)
__global__ __launch_bounds__(256) void k_gemm_dual_gelu_bf16(
    const bf16* __restrict__ A1, const bf16* __restrict__ Wt1,
    const bf16* __restrict__ A2, const bf16* __restrict__ Wt2,
    const float* __restrict__ bias, bf16* __restrict__ G,
    int M, int Nc, int K)
{
    const int lane = threadIdx.x & 31;
    const int wv   = blockIdx.x * (blockDim.x >> 5) + (threadIdx.x >> 5);
    const int nG = Nc >> 6, mT = M >> 4;
    if (wv >= mT * nG) return;
    const int m0 = (wv / nG) << 4;
    const int n0 = (wv % nG) << 6;
    const int colL = lane & 15;
    const int row  = m0 + colL;
    const int kA = (lane >> 4) << 4;
    const int kB = (lane >> 4) << 4;

    const bf16* a1p = A1 + (size_t)row * K + kA;
    const bf16* a2p = A2 + (size_t)row * K + kA;
    const bf16* b1p = Wt1 + (size_t)(n0 + colL) * K + kB;
    const bf16* b2p = Wt2 + (size_t)(n0 + colL) * K + kB;

    v8f acc0 = {}, acc1 = {}, acc2 = {}, acc3 = {};
    for (int kk = 0; kk < K; kk += 32) {
        v16bf a1 = ld_frag_a(a1p + kk);
        acc0 = wmma_bf16(a1, ld_frag_b(b1p + (size_t)(0 * 16) * K + kk), acc0);
        acc1 = wmma_bf16(a1, ld_frag_b(b1p + (size_t)(1 * 16) * K + kk), acc1);
        acc2 = wmma_bf16(a1, ld_frag_b(b1p + (size_t)(2 * 16) * K + kk), acc2);
        acc3 = wmma_bf16(a1, ld_frag_b(b1p + (size_t)(3 * 16) * K + kk), acc3);
        v16bf a2 = ld_frag_a(a2p + kk);
        acc0 = wmma_bf16(a2, ld_frag_b(b2p + (size_t)(0 * 16) * K + kk), acc0);
        acc1 = wmma_bf16(a2, ld_frag_b(b2p + (size_t)(1 * 16) * K + kk), acc1);
        acc2 = wmma_bf16(a2, ld_frag_b(b2p + (size_t)(2 * 16) * K + kk), acc2);
        acc3 = wmma_bf16(a2, ld_frag_b(b2p + (size_t)(3 * 16) * K + kk), acc3);
    }
    const int rbase = m0 + ((lane >> 4) << 3);
#pragma unroll
    for (int v = 0; v < 8; ++v) {
        const size_t r = (size_t)(rbase + v) * Nc;
        G[r + n0 +  0 + colL] = (bf16)gelu_exact(acc0[v] + bias[n0 +  0 + colL]);
        G[r + n0 + 16 + colL] = (bf16)gelu_exact(acc1[v] + bias[n0 + 16 + colL]);
        G[r + n0 + 32 + colL] = (bf16)gelu_exact(acc2[v] + bias[n0 + 32 + colL]);
        G[r + n0 + 48 + colL] = (bf16)gelu_exact(acc3[v] + bias[n0 + 48 + colL]);
    }
}

// ---------------------------------------------------------------------------
// BatchNorm column statistics: sum and sum-of-squares over rows.
// Private per-thread accumulators; one atomicAdd per block per column.
// Nc <= 512, blockDim = 256.
// ---------------------------------------------------------------------------
__global__ __launch_bounds__(256) void k_reduce_cols(
    const float* __restrict__ Y, float* __restrict__ sum, float* __restrict__ sq,
    int M, int Nc)
{
    const int tid = threadIdx.x;
    float s0 = 0.f, q0 = 0.f, s1 = 0.f, q1 = 0.f;
    for (int r = blockIdx.x; r < M; r += gridDim.x) {
        const float* row = Y + (size_t)r * Nc;
        if (tid < Nc)       { float v = row[tid];       s0 += v; q0 += v * v; }
        if (tid + 256 < Nc) { float v = row[tid + 256]; s1 += v; q1 += v * v; }
    }
    if (tid < Nc)       { atomicAdd(sum + tid,       s0); atomicAdd(sq + tid,       q0); }
    if (tid + 256 < Nc) { atomicAdd(sum + tid + 256, s1); atomicAdd(sq + tid + 256, q1); }
}

__global__ void k_bn_finalize(
    const float* __restrict__ sum, const float* __restrict__ sq,
    const float* __restrict__ gamma, const float* __restrict__ beta,
    float* __restrict__ scale, float* __restrict__ shift, int Nc, float invM)
{
    int c = blockIdx.x * blockDim.x + threadIdx.x;
    if (c >= Nc) return;
    float mean = sum[c] * invM;
    float var  = sq[c] * invM - mean * mean;    // biased, matches jnp.var
    float s    = gamma[c] * rsqrtf(var + BN_EPS);
    scale[c] = s;
    shift[c] = beta[c] - mean * s;
}

// ---------------------------------------------------------------------------
// Fused BN-apply epilogues
// ---------------------------------------------------------------------------
__global__ void k_apply_bn_f32_bf16(
    const float* __restrict__ Y, const float* __restrict__ scale,
    const float* __restrict__ shift, float* __restrict__ outF,
    bf16* __restrict__ outB, size_t n, int Nc)
{
    size_t i = (size_t)blockIdx.x * blockDim.x + threadIdx.x;
    if (i >= n) return;
    int c = (int)(i % (size_t)Nc);
    float v = Y[i] * scale[c] + shift[c];
    outF[i] = v;
    outB[i] = (bf16)v;
}

__global__ void k_apply_bn_res_f32_bf16(
    const float* __restrict__ Y, const float* __restrict__ scale,
    const float* __restrict__ shift, const float* __restrict__ res,
    float* __restrict__ outF, bf16* __restrict__ outB, size_t n, int Nc)
{
    size_t i = (size_t)blockIdx.x * blockDim.x + threadIdx.x;
    if (i >= n) return;
    int c = (int)(i % (size_t)Nc);
    float v = Y[i] * scale[c] + shift[c] + res[i];
    outF[i] = v;
    outB[i] = (bf16)v;
}

__global__ void k_apply_bn_gelu_bf16(
    const float* __restrict__ Y, const float* __restrict__ scale,
    const float* __restrict__ shift, bf16* __restrict__ outB, size_t n, int Nc)
{
    size_t i = (size_t)blockIdx.x * blockDim.x + threadIdx.x;
    if (i >= n) return;
    int c = (int)(i % (size_t)Nc);
    outB[i] = (bf16)gelu_exact(Y[i] * scale[c] + shift[c]);
}

__global__ void k_apply_bn_res_out(
    const float* __restrict__ Y, const float* __restrict__ scale,
    const float* __restrict__ shift, const float* __restrict__ res,
    float* __restrict__ out, size_t n, int Nc)
{
    size_t i = (size_t)blockIdx.x * blockDim.x + threadIdx.x;
    if (i >= n) return;
    int c = (int)(i % (size_t)Nc);
    out[i] = Y[i] * scale[c] + shift[c] + res[i];
}

// ---------------------------------------------------------------------------
// Edge scatter: agg[dst] += h[src]   (sum aggregation, GH=256 features)
// 8 threads per edge; thread p handles float4 lanes p, p+8, ... (coalesced).
// ---------------------------------------------------------------------------
__global__ __launch_bounds__(256) void k_scatter_sum(
    const long long* __restrict__ src, const long long* __restrict__ dst,
    const float* __restrict__ h, float* __restrict__ agg, int E)
{
    int t = blockIdx.x * blockDim.x + threadIdx.x;
    if (t >= E * 8) return;
    int e = t >> 3, p = t & 7;
    long long s = src[e], d = dst[e];
    const float4* hp = (const float4*)(h + (size_t)s * 256);
    float* ap = agg + (size_t)d * 256;
#pragma unroll
    for (int i = 0; i < 8; ++i) {
        float4 v = hp[i * 8 + p];
        int b = (i * 8 + p) * 4;
        atomicAdd(ap + b + 0, v.x);
        atomicAdd(ap + b + 1, v.y);
        atomicAdd(ap + b + 2, v.z);
        atomicAdd(ap + b + 3, v.w);
    }
}

// ---------------------------------------------------------------------------
// Host launcher
// ---------------------------------------------------------------------------
extern "C" void kernel_launch(void* const* d_in, const int* in_sizes, int n_in,
                              void* d_out, int out_size, void* d_ws, size_t ws_size,
                              hipStream_t stream) {
    (void)n_in; (void)out_size; (void)ws_size;

    const int N  = 100000;
    const int C  = 192;
    const int GH = 256;
    const int FH = 512;
    const int E  = in_sizes[1] / 2;

    // ---- inputs (setup_inputs order) ----
    const float*     x     = (const float*)d_in[0];
    const long long* ei    = (const long long*)d_in[1];   // int64 [2, E]
    const float*     Wg1   = (const float*)d_in[2];
    const float*     bg1   = (const float*)d_in[3];
    const float*     gg1   = (const float*)d_in[4];
    const float*     beg1  = (const float*)d_in[5];
    const float*     Wrel  = (const float*)d_in[6];
    const float*     brel  = (const float*)d_in[7];
    const float*     Wroot = (const float*)d_in[8];
    const float*     Wg2   = (const float*)d_in[9];
    const float*     bg2   = (const float*)d_in[10];
    const float*     gg2   = (const float*)d_in[11];
    const float*     beg2  = (const float*)d_in[12];
    const float*     Wf1   = (const float*)d_in[13];
    const float*     bf1   = (const float*)d_in[14];
    const float*     gf1   = (const float*)d_in[15];
    const float*     bef1  = (const float*)d_in[16];
    const float*     Wf2   = (const float*)d_in[17];
    const float*     bf2   = (const float*)d_in[18];
    const float*     gf2   = (const float*)d_in[19];
    const float*     bef2  = (const float*)d_in[20];
    float* out = (float*)d_out;
    const long long* esrc = ei;
    const long long* edst = ei + E;

    // ---- workspace carve-out (256B aligned) ----
    char* w = (char*)d_ws;
    auto carve = [&](size_t bytes) -> char* {
        char* p = w;
        w += (bytes + 255) & ~(size_t)255;
        return p;
    };
    bf16*  xb     = (bf16*) carve((size_t)N * C * 2);   // x in bf16
    float* y1     = (float*)carve((size_t)N * GH * 4);  // gemm1 out (reused: y2, y4)
    float* h_f32  = (float*)carve((size_t)N * GH * 4);  // BN(h), f32 (y3 overlays here)
    bf16*  h_b    = (bf16*) carve((size_t)N * GH * 2);  // BN(h), bf16
    float* agg    = (float*)carve((size_t)N * GH * 4);  // scatter accumulator
    bf16*  agg_b  = (bf16*) carve((size_t)N * GH * 2);
    bf16*  g_b    = (bf16*) carve((size_t)N * GH * 2);  // gelu(graphconv) bf16
    float* x1     = (float*)carve((size_t)N * C * 4);   // grapher output (residual)
    bf16*  x1b    = (bf16*) carve((size_t)N * C * 2);
    bf16*  t3     = (bf16*) carve((size_t)N * FH * 2);  // gelu(BN(ffn fc1)) bf16
    bf16*  Wtg1   = (bf16*) carve((size_t)C * GH * 2);  // transposed bf16 weights
    bf16*  Wtrel  = (bf16*) carve((size_t)GH * GH * 2);
    bf16*  Wtroot = (bf16*) carve((size_t)GH * GH * 2);
    bf16*  Wtg2   = (bf16*) carve((size_t)GH * C * 2);
    bf16*  Wtf1   = (bf16*) carve((size_t)C * FH * 2);
    bf16*  Wtf2   = (bf16*) carve((size_t)FH * C * 2);
    float* st     = (float*)carve((size_t)16 * 512 * 4); // stats: 16 arrays of 512
    // stat slots: sum[l]=st+l*512, sq[l]=st+(4+l)*512, scale=st+(8+l)*512, shift=st+(12+l)*512
    auto ssum   = [&](int l) { return st + (size_t)l * 512; };
    auto ssq    = [&](int l) { return st + (size_t)(4 + l) * 512; };
    auto sscale = [&](int l) { return st + (size_t)(8 + l) * 512; };
    auto sshift = [&](int l) { return st + (size_t)(12 + l) * 512; };
    // buffer aliases (dead-region reuse)
    float* y2 = y1;
    float* y3 = h_f32;   // N*FH*4 = 204.8MB fits in h_f32+h_b+agg region (dead by then)
    float* y4 = y1;

    const float invN = 1.0f / (float)N;
    auto eb = [](size_t n) { return (unsigned)((n + 255) / 256); };  // elementwise blocks
    auto gb = [](int M, int Nc) {                                    // gemm blocks
        int waves = (M >> 4) * (Nc >> 6);
        return (unsigned)((waves + 7) / 8);
    };

    // ---- 0. zero accumulators (deterministic per call) ----
    k_zero_f32<<<2048, 256, 0, stream>>>(agg, (size_t)N * GH);
    k_zero_f32<<<16, 256, 0, stream>>>(st, (size_t)8 * 512);

    // ---- 1. precision prep: x -> bf16, weights -> bf16 transposed ----
    k_f32_to_bf16<<<eb((size_t)N * C), 256, 0, stream>>>(x, xb, (size_t)N * C);
    k_transpose_w<<<eb((size_t)C * GH),  256, 0, stream>>>(Wg1,   Wtg1,   C,  GH);
    k_transpose_w<<<eb((size_t)GH * GH), 256, 0, stream>>>(Wrel,  Wtrel,  GH, GH);
    k_transpose_w<<<eb((size_t)GH * GH), 256, 0, stream>>>(Wroot, Wtroot, GH, GH);
    k_transpose_w<<<eb((size_t)GH * C),  256, 0, stream>>>(Wg2,   Wtg2,   GH, C);
    k_transpose_w<<<eb((size_t)C * FH),  256, 0, stream>>>(Wf1,   Wtf1,   C,  FH);
    k_transpose_w<<<eb((size_t)FH * C),  256, 0, stream>>>(Wf2,   Wtf2,   FH, C);

    // ---- 2. Grapher fc1: y1 = x @ Wg1 + bg1 ; BN stats ; h = BN(y1) ----
    k_gemm_bias_f32<<<gb(N, GH), 256, 0, stream>>>(xb, Wtg1, bg1, y1, N, GH, C);
    k_reduce_cols<<<512, 256, 0, stream>>>(y1, ssum(0), ssq(0), N, GH);
    k_bn_finalize<<<2, 256, 0, stream>>>(ssum(0), ssq(0), gg1, beg1, sscale(0), sshift(0), GH, invN);
    k_apply_bn_f32_bf16<<<eb((size_t)N * GH), 256, 0, stream>>>(
        y1, sscale(0), sshift(0), h_f32, h_b, (size_t)N * GH, GH);

    // ---- 3. GraphConv aggregate: agg[dst] += h[src] ----
    k_scatter_sum<<<(unsigned)((E * 8 + 255) / 256), 256, 0, stream>>>(esrc, edst, h_f32, agg, E);
    k_f32_to_bf16<<<eb((size_t)N * GH), 256, 0, stream>>>(agg, agg_b, (size_t)N * GH);

    // ---- 4. GraphConv linear + GELU: g = gelu(agg@Wrel + h@Wroot + brel) ----
    k_gemm_dual_gelu_bf16<<<gb(N, GH), 256, 0, stream>>>(
        agg_b, Wtrel, h_b, Wtroot, brel, g_b, N, GH, GH);

    // ---- 5. Grapher fc2 + BN + residual: x1 = BN(g @ Wg2 + bg2) + x ----
    k_gemm_bias_f32<<<gb(N, C), 256, 0, stream>>>(g_b, Wtg2, bg2, y2, N, C, GH);
    k_reduce_cols<<<512, 256, 0, stream>>>(y2, ssum(1), ssq(1), N, C);
    k_bn_finalize<<<1, 256, 0, stream>>>(ssum(1), ssq(1), gg2, beg2, sscale(1), sshift(1), C, invN);
    k_apply_bn_res_f32_bf16<<<eb((size_t)N * C), 256, 0, stream>>>(
        y2, sscale(1), sshift(1), x, x1, x1b, (size_t)N * C, C);

    // ---- 6. FFN fc1 + BN + GELU: t3 = gelu(BN(x1 @ Wf1 + bf1)) ----
    k_gemm_bias_f32<<<gb(N, FH), 256, 0, stream>>>(x1b, Wtf1, bf1, y3, N, FH, C);
    k_reduce_cols<<<512, 256, 0, stream>>>(y3, ssum(2), ssq(2), N, FH);
    k_bn_finalize<<<2, 256, 0, stream>>>(ssum(2), ssq(2), gf1, bef1, sscale(2), sshift(2), FH, invN);
    k_apply_bn_gelu_bf16<<<eb((size_t)N * FH), 256, 0, stream>>>(
        y3, sscale(2), sshift(2), t3, (size_t)N * FH, FH);

    // ---- 7. FFN fc2 + BN + residual: out = BN(t3 @ Wf2 + bf2) + x1 ----
    k_gemm_bias_f32<<<gb(N, C), 256, 0, stream>>>(t3, Wtf2, bf2, y4, N, C, FH);
    k_reduce_cols<<<512, 256, 0, stream>>>(y4, ssum(3), ssq(3), N, C);
    k_bn_finalize<<<1, 256, 0, stream>>>(ssum(3), ssq(3), gf2, bef2, sscale(3), sshift(3), C, invN);
    k_apply_bn_res_out<<<eb((size_t)N * C), 256, 0, stream>>>(
        y4, sscale(3), sshift(3), x1, out, (size_t)N * C, C);
}
